// LuongAttnDecoderRNN_47605417509048
// MI455X (gfx1250) — matrix operand order repaired and checked by reading, not compiled, not run on hardware
//
#include <hip/hip_runtime.h>
#include <hip/hip_bf16.h>

// ---------------------------------------------------------------------------
// LuongAttnDecoderRNN single step, CDNA5 (gfx1250) wave32 + bf16 WMMA + TDM.
// B=64, L=256, H=E=1024, V=50257.
// ---------------------------------------------------------------------------

#define B_ 64
#define L_ 256
#define H_ 1024
#define V_ 50257
#define G3_ 3072
#define KCH 128   // K elements per TDM chunk (512 B per row, 16 rows = 8 KB)

typedef __attribute__((ext_vector_type(16))) __bf16 v16bf;
typedef __attribute__((ext_vector_type(8)))  float  v8f;
typedef unsigned int u32x4 __attribute__((ext_vector_type(4)));
typedef int          i32x4 __attribute__((ext_vector_type(4)));
typedef int          i32x8 __attribute__((ext_vector_type(8)));

__device__ __forceinline__ unsigned short f2us(float f) {
  union { float f; unsigned u; } v; v.f = f;
  unsigned u = v.u + 0x7FFFu + ((v.u >> 16) & 1u);   // round-to-nearest-even
  return (unsigned short)(u >> 16);
}
__device__ __forceinline__ __bf16 us2bf(unsigned short s) {
  union { unsigned short u; __bf16 b; } x; x.u = s; return x.b;
}
__device__ __forceinline__ __bf16 f2bf(float f) { return us2bf(f2us(f)); }

// ---------------------------------------------------------------------------
// Tensor Data Mover: 2D tile load (Global -> LDS), D# per CDNA5 ISA §8.
// ---------------------------------------------------------------------------
#if __has_builtin(__builtin_amdgcn_tensor_load_to_lds) && \
    __has_builtin(__builtin_amdgcn_s_wait_tensorcnt)
#define USE_TDM 1
#else
#define USE_TDM 0
#endif

#if USE_TDM
__device__ __forceinline__ unsigned lds_off(const void* p) {
  // generic LDS address: low 32 bits are the LDS byte offset (ISA §10.2)
  return (unsigned)(unsigned long long)(uintptr_t)p;
}

// 2D tile: tile_w elements (4B each) x tile_h rows; row stride in elements.
// Rows beyond tensor_h read back as zero (OOB policy) -> natural V-tail.
__device__ __forceinline__ void tdm_load_2d(unsigned lds_byte,
                                            const void* gaddr,
                                            unsigned tile_w, unsigned tile_h,
                                            unsigned tensor_w, unsigned tensor_h,
                                            unsigned row_stride_elems) {
  unsigned long long ga = (unsigned long long)(uintptr_t)gaddr;
  u32x4 g0;
  g0[0] = 1u;                                        // count=1, user descriptor
  g0[1] = lds_byte;                                  // lds_addr
  g0[2] = (unsigned)(ga & 0xffffffffu);              // global_addr[31:0]
  g0[3] = (unsigned)((ga >> 32) & 0x01ffffffu)       // global_addr[56:32]
          | (2u << 30);                              // type = 2 ("image")
  i32x8 g1;
  g1[0] = (int)(2u << 16);                           // data_size=2 (4B), mask=0
  g1[1] = (int)((tensor_w & 0xffffu) << 16);         // tensor_dim0[15:0]
  g1[2] = (int)(((tensor_w >> 16) & 0xffffu) |
                ((tensor_h & 0xffffu) << 16));       // dim0[31:16] | dim1[15:0]
  g1[3] = (int)(((tensor_h >> 16) & 0xffffu) |
                ((tile_w & 0xffffu) << 16));         // dim1[31:16] | tile_dim0
  g1[4] = (int)(tile_h & 0xffffu);                   // tile_dim1 (tile_dim2=0)
  g1[5] = (int)row_stride_elems;                     // tensor_dim0_stride[31:0]
  g1[6] = 0;                                         // stride[47:32] | dim1_stride
  g1[7] = 0;
  i32x4 z4 = {0, 0, 0, 0};
#if defined(__clang_major__) && (__clang_major__ >= 23)
  i32x8 z8 = {0, 0, 0, 0, 0, 0, 0, 0};
  __builtin_amdgcn_tensor_load_to_lds(g0, g1, z4, z4, z8, 0);
#else
  __builtin_amdgcn_tensor_load_to_lds(g0, g1, z4, z4, 0);
#endif
}
#endif  // USE_TDM

// ---------------------------------------------------------------------------
// K1: Gi = emb[seq] @ w_ih^T + b_ih ; Gh = h0 @ w_hh^T + b_hh   (both [64,3072])
// grid (4, 24), block 256 (8 waves; wave = one 16x16 N-tile, 128 N per block)
// ---------------------------------------------------------------------------
__global__ __launch_bounds__(256) void k_gru_gates(
    const int* __restrict__ seq, const float* __restrict__ h0,
    const float* __restrict__ emb, const float* __restrict__ w_ih,
    const float* __restrict__ w_hh, const float* __restrict__ b_ih,
    const float* __restrict__ b_hh, float* __restrict__ Gi,
    float* __restrict__ Gh) {
  __shared__ unsigned short Ax[16 * H_];   // x tile, bf16
  __shared__ unsigned short Ah[16 * H_];   // h tile, bf16
  const int tid = threadIdx.x;
  const int mb  = blockIdx.x * 16;

  for (int idx = tid; idx < 16 * H_; idx += 256) {
    int r = idx >> 10, k = idx & (H_ - 1);
    int brow = mb + r;
    Ax[idx] = f2us(emb[(long long)seq[brow] * H_ + k]);
    Ah[idx] = f2us(h0[brow * H_ + k]);
  }
  __syncthreads();

  const int wave = tid >> 5, lane = tid & 31;
  const int nb  = blockIdx.y * 128 + wave * 16;
  const int am  = lane & 15;            // A row / B column / D column
  const int hi  = (lane >> 4) & 1;      // lane-half selector
  const float* wi_row = w_ih + (long long)(nb + am) * H_;
  const float* wh_row = w_hh + (long long)(nb + am) * H_;

  v8f ci = {}; v8f ch = {};
  for (int kk = 0; kk < H_; kk += 32) {
    v16bf a, b;
#pragma unroll
    for (int j = 0; j < 8; ++j) {        // ISA 16-bit A 16x32 layout
      a[j]     = us2bf(Ax[am * H_ + kk + hi * 8 + j]);
      a[j + 8] = us2bf(Ax[am * H_ + kk + 16 + hi * 8 + j]);
    }
#pragma unroll
    for (int j = 0; j < 16; ++j)         // ISA 16-bit B 32x16 layout
      b[j] = f2bf(wi_row[kk + hi * 8 + (j & 7) + (j >> 3) * 16]);
    ci = __builtin_amdgcn_wmma_f32_16x16x32_bf16(false, a, false, b,
                                                 (short)0, ci, false, false);
#pragma unroll
    for (int j = 0; j < 8; ++j) {
      a[j]     = us2bf(Ah[am * H_ + kk + hi * 8 + j]);
      a[j + 8] = us2bf(Ah[am * H_ + kk + 16 + hi * 8 + j]);
    }
#pragma unroll
    for (int j = 0; j < 16; ++j)
      b[j] = f2bf(wh_row[kk + hi * 8 + (j & 7) + (j >> 3) * 16]);
    ch = __builtin_amdgcn_wmma_f32_16x16x32_bf16(false, a, false, b,
                                                 (short)0, ch, false, false);
  }

  const int n = nb + am;
  const float bi = b_ih[n], bh = b_hh[n];
#pragma unroll
  for (int r = 0; r < 8; ++r) {          // C/D layout: m = r + hi*8, n = lane&15
    int m = mb + r + hi * 8;
    Gi[(long long)m * G3_ + n] = ci[r] + bi;
    Gh[(long long)m * G3_ + n] = ch[r] + bh;
  }
}

// ---------------------------------------------------------------------------
// K2: GRU combine -> h_new [64,1024]; also writes `hidden` output slice.
// ---------------------------------------------------------------------------
__global__ __launch_bounds__(256) void k_gru_combine(
    const float* __restrict__ Gi, const float* __restrict__ Gh,
    const float* __restrict__ h0, float* __restrict__ h_new,
    float* __restrict__ hid_out) {
  int t = blockIdx.x * blockDim.x + threadIdx.x;   // 65536 threads
  int b = t >> 10, j = t & (H_ - 1);
  long long base = (long long)b * G3_;
  float ir = Gi[base + j],        hr = Gh[base + j];
  float iz = Gi[base + H_ + j],   hz = Gh[base + H_ + j];
  float in_ = Gi[base + 2 * H_ + j], hn = Gh[base + 2 * H_ + j];
  float r = 1.f / (1.f + expf(-(ir + hr)));
  float z = 1.f / (1.f + expf(-(iz + hz)));
  float n = tanhf(in_ + r * hn);
  float hv = (1.f - z) * n + z * h0[t];
  h_new[t]   = hv;
  hid_out[t] = hv;
}

// ---------------------------------------------------------------------------
// K3: q[b,h] = sum_g h_new[b,g] * attn_w[g,h]   (64x1024x1024, WMMA)
// grid (4, 8), block 256.  B operand is column-strided in attn_w.
// ---------------------------------------------------------------------------
__global__ __launch_bounds__(256) void k_attn_q(
    const float* __restrict__ h_new, const float* __restrict__ attn_w,
    float* __restrict__ q) {
  __shared__ unsigned short As[16 * H_];
  const int tid = threadIdx.x;
  const int mb  = blockIdx.x * 16;
  for (int idx = tid; idx < 16 * H_; idx += 256) {
    int r = idx >> 10, k = idx & (H_ - 1);
    As[idx] = f2us(h_new[(mb + r) * H_ + k]);
  }
  __syncthreads();

  const int wave = tid >> 5, lane = tid & 31;
  const int nb = blockIdx.y * 128 + wave * 16;
  const int am = lane & 15, hi = (lane >> 4) & 1;
  const int n  = nb + am;

  v8f acc = {};
  for (int kk = 0; kk < H_; kk += 32) {
    v16bf a, b;
#pragma unroll
    for (int j = 0; j < 8; ++j) {
      a[j]     = us2bf(As[am * H_ + kk + hi * 8 + j]);
      a[j + 8] = us2bf(As[am * H_ + kk + 16 + hi * 8 + j]);
    }
#pragma unroll
    for (int j = 0; j < 16; ++j)
      b[j] = f2bf(attn_w[(long long)(kk + hi * 8 + (j & 7) + (j >> 3) * 16) * H_ + n]);
    acc = __builtin_amdgcn_wmma_f32_16x16x32_bf16(false, a, false, b,
                                                  (short)0, acc, false, false);
  }
#pragma unroll
  for (int r = 0; r < 8; ++r)
    q[(long long)(mb + r + hi * 8) * H_ + n] = acc[r];
}

// ---------------------------------------------------------------------------
// K4: per-batch scores -> softmax -> attn_weights + context.
// grid (64), block 256.  scores[b,l] = q[b]·enc[l,b] + h_new[b]·attn_b.
// ---------------------------------------------------------------------------
__global__ __launch_bounds__(256) void k_attn_softmax_ctx(
    const float* __restrict__ q, const float* __restrict__ h_new,
    const float* __restrict__ attn_b, const float* __restrict__ enc,
    float* __restrict__ attn_out, float* __restrict__ context) {
  __shared__ float qs[H_];
  __shared__ float sc[L_];
  __shared__ float red[8];
  const int b = blockIdx.x, tid = threadIdx.x;
  const int wave = tid >> 5, lane = tid & 31;

  float p = 0.f;
  for (int j = tid; j < H_; j += 256) {
    float qv = q[b * H_ + j];
    qs[j] = qv;
    p += h_new[b * H_ + j] * attn_b[j];
  }
  for (int o = 16; o; o >>= 1) p += __shfl_down(p, o, 32);
  if (lane == 0) red[wave] = p;
  __syncthreads();
  float sb = 0.f;
#pragma unroll
  for (int i = 0; i < 8; ++i) sb += red[i];

  // one wave sweeps 32 l values; lanes split H (coalesced)
  for (int t = 0; t < 32; ++t) {
    int l = wave * 32 + t;
    const float* e = enc + (long long)l * (B_ * H_) + b * H_;
    float s = 0.f;
    for (int h = lane; h < H_; h += 32) s += qs[h] * e[h];
    for (int o = 16; o; o >>= 1) s += __shfl_down(s, o, 32);
    if (lane == 0) sc[l] = s + sb;
  }
  __syncthreads();

  // softmax over L_ = 256 (one score per thread)
  float sv = sc[tid];
  float m = sv;
  for (int o = 16; o; o >>= 1) m = fmaxf(m, __shfl_down(m, o, 32));
  __syncthreads();
  if (lane == 0) red[wave] = m;
  __syncthreads();
  m = red[0];
#pragma unroll
  for (int i = 1; i < 8; ++i) m = fmaxf(m, red[i]);
  float ev = expf(sv - m);
  float s = ev;
  for (int o = 16; o; o >>= 1) s += __shfl_down(s, o, 32);
  __syncthreads();
  if (lane == 0) red[wave] = s;
  __syncthreads();
  s = 0.f;
#pragma unroll
  for (int i = 0; i < 8; ++i) s += red[i];
  float w = ev / s;
  sc[tid] = w;
  attn_out[b * L_ + tid] = w;
  __syncthreads();

  // context[b,h] = sum_l w[l] * enc[l,b,h]
  for (int h = tid; h < H_; h += 256) {
    float acc = 0.f;
    for (int l = 0; l < L_; ++l)
      acc += sc[l] * enc[(long long)l * (B_ * H_) + b * H_ + h];
    context[b * H_ + h] = acc;
  }
}

// ---------------------------------------------------------------------------
// K5: concat_out = tanh([h_new | context] @ concat_w^T + concat_b)  (K = 2048)
// grid (4, 8), block 256.
// ---------------------------------------------------------------------------
__global__ __launch_bounds__(256) void k_concat_tanh(
    const float* __restrict__ h_new, const float* __restrict__ context,
    const float* __restrict__ concat_w, const float* __restrict__ concat_b,
    float* __restrict__ concat_out) {
  __shared__ unsigned short As[16 * 2048];   // 64 KB
  const int tid = threadIdx.x;
  const int mb  = blockIdx.x * 16;
  for (int idx = tid; idx < 16 * 2048; idx += 256) {
    int r = idx >> 11, k = idx & 2047;
    float v = (k < H_) ? h_new[(mb + r) * H_ + k]
                       : context[(mb + r) * H_ + (k - H_)];
    As[idx] = f2us(v);
  }
  __syncthreads();

  const int wave = tid >> 5, lane = tid & 31;
  const int nb = blockIdx.y * 128 + wave * 16;
  const int am = lane & 15, hi = (lane >> 4) & 1;
  const int n  = nb + am;
  const float* wr = concat_w + (long long)n * 2048;

  v8f acc = {};
  for (int kk = 0; kk < 2048; kk += 32) {
    v16bf a, b;
#pragma unroll
    for (int j = 0; j < 8; ++j) {
      a[j]     = us2bf(As[am * 2048 + kk + hi * 8 + j]);
      a[j + 8] = us2bf(As[am * 2048 + kk + 16 + hi * 8 + j]);
    }
#pragma unroll
    for (int j = 0; j < 16; ++j)
      b[j] = f2bf(wr[kk + hi * 8 + (j & 7) + (j >> 3) * 16]);
    acc = __builtin_amdgcn_wmma_f32_16x16x32_bf16(false, a, false, b,
                                                  (short)0, acc, false, false);
  }
  const float cb = concat_b[n];
#pragma unroll
  for (int r = 0; r < 8; ++r)
    concat_out[(long long)(mb + r + hi * 8) * H_ + n] = tanhf(acc[r] + cb);
}

// ---------------------------------------------------------------------------
// K6: output = concat_out @ out_w^T + out_b   [64, 50257]   (206 MB stream)
// grid (4, 393), block 256.  Weight tiles arrive via TDM (double-buffered
// tensor_load_to_lds + s_wait_tensorcnt); OOB rows read as zero, stores
// tail-guarded at V.
// ---------------------------------------------------------------------------
__global__ __launch_bounds__(256) void k_out_gemm(
    const float* __restrict__ concat_out, const float* __restrict__ out_w,
    const float* __restrict__ out_b, float* __restrict__ out) {
  __shared__ unsigned short As[16 * H_];        // 32 KB (A tile, bf16)
#if USE_TDM
  __shared__ float Bs[8][2][16 * KCH];          // 8 waves x 2 bufs x 8 KB = 128 KB
#endif
  const int tid = threadIdx.x;
  const int mb  = blockIdx.x * 16;
  for (int idx = tid; idx < 16 * H_; idx += 256) {
    int r = idx >> 10, k = idx & (H_ - 1);
    As[idx] = f2us(concat_out[(mb + r) * H_ + k]);
  }
  __syncthreads();

  const int wave = tid >> 5, lane = tid & 31;
  const int nb = blockIdx.y * 128 + wave * 16;
  const int am = lane & 15, hi = (lane >> 4) & 1;
  const int n  = nb + am;

  v8f acc = {};
#if USE_TDM
  if (nb < V_) {
    const unsigned rows_left = (unsigned)(V_ - nb);   // OOB rows -> zeros

    // one WMMA sweep over a staged KCH chunk
    auto consume = [&](const float* bp, int c) {
#pragma unroll
      for (int kc = 0; kc < KCH; kc += 32) {
        const int kk = c * KCH + kc;
        v16bf a, b;
#pragma unroll
        for (int j = 0; j < 8; ++j) {
          a[j]     = us2bf(As[am * H_ + kk + hi * 8 + j]);
          a[j + 8] = us2bf(As[am * H_ + kk + 16 + hi * 8 + j]);
        }
#pragma unroll
        for (int j = 0; j < 16; ++j)
          b[j] = f2bf(bp[am * KCH + kc + hi * 8 + (j & 7) + (j >> 3) * 16]);
        acc = __builtin_amdgcn_wmma_f32_16x16x32_bf16(false, a, false, b,
                                                      (short)0, acc, false,
                                                      false);
      }
    };

    // prologue: chunk 0 in flight
    tdm_load_2d(lds_off(&Bs[wave][0][0]), out_w + (long long)nb * H_,
                KCH, 16, KCH, rows_left, H_);
    // steady state: issue c+1, wait for c (<=1 outstanding), consume c
#pragma unroll
    for (int c = 0; c < H_ / KCH - 1; ++c) {          // 7 iters
      tdm_load_2d(lds_off(&Bs[wave][(c + 1) & 1][0]),
                  out_w + (long long)nb * H_ + (c + 1) * KCH,
                  KCH, 16, KCH, rows_left, H_);
      __builtin_amdgcn_s_wait_tensorcnt(1);           // literal immediate
      consume(&Bs[wave][c & 1][0], c);
    }
    // epilogue: drain and consume last chunk
    __builtin_amdgcn_s_wait_tensorcnt(0);             // literal immediate
    consume(&Bs[wave][(H_ / KCH - 1) & 1][0], H_ / KCH - 1);
  }
#else
  {
    const int rw = (n < V_) ? n : (V_ - 1);
    const float* wr = out_w + (long long)rw * H_;
    for (int kk = 0; kk < H_; kk += 32) {
      __builtin_prefetch(wr + kk + 256, 0, 0);
      v16bf a, b;
#pragma unroll
      for (int j = 0; j < 8; ++j) {
        a[j]     = us2bf(As[am * H_ + kk + hi * 8 + j]);
        a[j + 8] = us2bf(As[am * H_ + kk + 16 + hi * 8 + j]);
      }
#pragma unroll
      for (int j = 0; j < 16; ++j)
        b[j] = f2bf(wr[kk + hi * 8 + (j & 7) + (j >> 3) * 16]);
      acc = __builtin_amdgcn_wmma_f32_16x16x32_bf16(false, a, false, b,
                                                    (short)0, acc, false,
                                                    false);
    }
  }
#endif
  if (n < V_) {
    const float bias = out_b[n];
#pragma unroll
    for (int r = 0; r < 8; ++r)
      out[(long long)(mb + r + hi * 8) * V_ + n] = acc[r] + bias;
  }
}

// ---------------------------------------------------------------------------
extern "C" void kernel_launch(void* const* d_in, const int* in_sizes, int n_in,
                              void* d_out, int out_size, void* d_ws,
                              size_t ws_size, hipStream_t stream) {
  (void)in_sizes; (void)n_in; (void)out_size; (void)ws_size;
  const int*   seq      = (const int*)d_in[0];
  const float* h0       = (const float*)d_in[1];   // [1,B,H]
  const float* enc      = (const float*)d_in[2];   // [L,B,H]
  const float* emb      = (const float*)d_in[3];   // [V,E]
  const float* w_ih     = (const float*)d_in[4];   // [3H,E]
  const float* w_hh     = (const float*)d_in[5];   // [3H,H]
  const float* b_ih     = (const float*)d_in[6];
  const float* b_hh     = (const float*)d_in[7];
  const float* attn_w   = (const float*)d_in[8];   // [H,H]
  const float* attn_b   = (const float*)d_in[9];
  const float* concat_w = (const float*)d_in[10];  // [H,2H]
  const float* concat_b = (const float*)d_in[11];
  const float* out_w    = (const float*)d_in[12];  // [V,H]
  const float* out_b    = (const float*)d_in[13];

  float* out      = (float*)d_out;                       // [B,V]
  float* hid_out  = out + (long long)B_ * V_;            // [1,B,H]
  float* attn_out = hid_out + (long long)B_ * H_;        // [B,1,L]

  float* ws        = (float*)d_ws;
  float* Gi        = ws;                                 // [B,3H]
  float* Gh        = Gi + (long long)B_ * G3_;           // [B,3H]
  float* h_new     = Gh + (long long)B_ * G3_;           // [B,H]
  float* q         = h_new + (long long)B_ * H_;         // [B,H]
  float* context   = q + (long long)B_ * H_;             // [B,H]
  float* concat_o  = context + (long long)B_ * H_;       // [B,H]

  k_gru_gates<<<dim3(4, 24), 256, 0, stream>>>(seq, h0, emb, w_ih, w_hh,
                                               b_ih, b_hh, Gi, Gh);
  k_gru_combine<<<dim3(256), 256, 0, stream>>>(Gi, Gh, h0, h_new, hid_out);
  k_attn_q<<<dim3(4, 8), 256, 0, stream>>>(h_new, attn_w, q);
  k_attn_softmax_ctx<<<dim3(64), 256, 0, stream>>>(q, h_new, attn_b, enc,
                                                   attn_out, context);
  k_concat_tanh<<<dim3(4, 8), 256, 0, stream>>>(h_new, context, concat_w,
                                                concat_b, concat_o);
  k_out_gemm<<<dim3(4, 393), 256, 0, stream>>>(concat_o, out_w, out_b, out);
}
